// LSTM_55344948576590
// MI455X (gfx1250) — compile-verified, matrix operand order checked by reading, not run on hardware
//
#include <hip/hip_runtime.h>
#include <hip/hip_bf16.h>

// ---------------------------------------------------------------------------
// Types for CDNA5 WMMA (wave32): 16x16x32 f16 -> f32
// ---------------------------------------------------------------------------
typedef _Float16 f16;
typedef __attribute__((ext_vector_type(16))) _Float16 v16h;
typedef __attribute__((ext_vector_type(8)))  _Float16 v8h;
typedef __attribute__((ext_vector_type(8)))  float    v8f;

#define BATCH 512
#define TSEQ  256
#define CNNC  512
#define VOC   37
#define HID   8
#define KPAD  64   // V padded for WMMA K
#define GN    32   // 4*H gate width
#define NROW  (TSEQ * BATCH)      // 131072 sequence rows
#define NTILE (NROW / 16)         // 8192 16-row tiles

__device__ __forceinline__ v8f wmma16(v16h a, v16h b, v8f c) {
  return __builtin_amdgcn_wmma_f32_16x16x32_f16(
      /*neg_a=*/false, a, /*neg_b=*/false, b,
      /*c_mod=*/(short)0, c, /*reuse_a=*/false, /*reuse_b=*/false);
}

__device__ __forceinline__ v16h cat8(v8h a, v8h b) {
  v16h r;
  for (int i = 0; i < 8; ++i) { r[i] = a[i]; r[i + 8] = b[i]; }
  return r;
}

__device__ __forceinline__ v8f splat8(float x) {
  v8f r;
  for (int i = 0; i < 8; ++i) r[i] = x;
  return r;
}

__device__ __forceinline__ float sigmoidf_(float x) {
  return 1.0f / (1.0f + __expf(-x));
}
__device__ __forceinline__ float tanhf_(float x) {
  float t = __expf(-2.0f * fabsf(x));
  float r = (1.0f - t) / (1.0f + t);
  return copysignf(r, x);
}

// ---------------------------------------------------------------------------
// K1: adaptive avg pool -> pooled_h f16 [B][CNN]
// ---------------------------------------------------------------------------
__global__ void lstm_pool_kernel(const float* __restrict__ features,
                                 f16* __restrict__ pooled) {
  int idx = blockIdx.x * blockDim.x + threadIdx.x;   // over B*CNN
  if (idx >= BATCH * CNNC) return;
  const float* p = features + (size_t)idx * 49;
  float s = 0.f;
  #pragma unroll
  for (int i = 0; i < 49; ++i) s += p[i];
  pooled[idx] = (f16)(s * (1.0f / 49.0f));
}

// ---------------------------------------------------------------------------
// K2: pack weights into gfx1250 WMMA B-operand lane layout (ISA 7.12.2):
//   per tile (32x16 KxN f16): halfs[lane*16 + h], element h<8 -> K = kb+h,
//   h>=8 -> K = kb+16+(h-8), with kb = (lane/16)*8; lane%16 = N.
// ---------------------------------------------------------------------------
__global__ void lstm_pack_kernel(const float* __restrict__ W_in,
                                 const float* __restrict__ W_ih,
                                 const float* __restrict__ W_hh,
                                 const float* __restrict__ W_out,
                                 const float* __restrict__ b_ih,
                                 const float* __restrict__ b_hh,
                                 const float* __restrict__ b_out,
                                 f16* __restrict__ Winp,   // [48][512]
                                 f16* __restrict__ Wihp,   // 4 tiles (ks*2+nt)
                                 f16* __restrict__ Whhp,   // 2 tiles
                                 f16* __restrict__ Woutp,  // 3 tiles
                                 float* __restrict__ gbias, // [32]
                                 float* __restrict__ obias) // [48]
{
  int tid = threadIdx.x;  // 256 threads, 1 block
  for (int idx = tid; idx < 48 * CNNC; idx += 256) {
    int n = idx / CNNC, k = idx % CNNC;
    Winp[idx] = (f16)(n < VOC ? W_in[n * CNNC + k] : 0.f);
  }
  for (int idx = tid; idx < 4 * 512; idx += 256) {  // W_ih [32][37]
    int tile = idx / 512, e = idx % 512, lane = e / 16, h = e % 16;
    int ks = tile / 2, nt = tile % 2;
    int kl = (h < 8 ? h : h + 8) + (lane / 16) * 8;
    int k = ks * 32 + kl;
    int n = nt * 16 + (lane % 16);
    Wihp[idx] = (f16)(k < VOC ? W_ih[n * VOC + k] : 0.f);
  }
  for (int idx = tid; idx < 2 * 512; idx += 256) {  // W_hh [32][8]
    int tile = idx / 512, e = idx % 512, lane = e / 16, h = e % 16;
    int kl = (h < 8 ? h : h + 8) + (lane / 16) * 8;
    int n = tile * 16 + (lane % 16);
    Whhp[idx] = (f16)(kl < HID ? W_hh[n * HID + kl] : 0.f);
  }
  for (int idx = tid; idx < 3 * 512; idx += 256) {  // W_out [37][8]
    int tile = idx / 512, e = idx % 512, lane = e / 16, h = e % 16;
    int kl = (h < 8 ? h : h + 8) + (lane / 16) * 8;
    int n = tile * 16 + (lane % 16);
    Woutp[idx] = (f16)((kl < HID && n < VOC) ? W_out[n * HID + kl] : 0.f);
  }
  if (tid < GN) gbias[tid] = b_ih[tid] + b_hh[tid];
  if (tid < 48) obias[tid] = (tid < VOC) ? b_out[tid] : 0.f;
}

// ---------------------------------------------------------------------------
// K3: feat = pooled @ W_in^T + b_in  -> Xh row (t=0), f16, padded to KPAD.
// One wave per 16-row M tile. M=512, N=48 (3 tiles), K=512 (16 k-steps).
// ---------------------------------------------------------------------------
__global__ void lstm_featgemm_kernel(const f16* __restrict__ pooled,
                                     const f16* __restrict__ Winp,
                                     const float* __restrict__ b_in,
                                     f16* __restrict__ Xh) {
  const int L = threadIdx.x;           // 32 lanes
  const int j16 = L & 15, half = L >> 4;
  const int Mb = blockIdx.x * 16;

  v8f acc[3];
  #pragma unroll
  for (int nt = 0; nt < 3; ++nt) {
    int col = nt * 16 + j16;
    acc[nt] = splat8(col < VOC ? b_in[col] : 0.f);
  }

  for (int kt = 0; kt < CNNC / 32; ++kt) {
    int kb = kt * 32 + half * 8;
    int row = Mb + j16;
    v16h A = cat8(*(const v8h*)(pooled + (size_t)row * CNNC + kb),
                  *(const v8h*)(pooled + (size_t)row * CNNC + kb + 16));
    #pragma unroll
    for (int nt = 0; nt < 3; ++nt) {
      int n = nt * 16 + j16;
      v16h Bm = cat8(*(const v8h*)(Winp + (size_t)n * CNNC + kb),
                     *(const v8h*)(Winp + (size_t)n * CNNC + kb + 16));
      acc[nt] = wmma16(A, Bm, acc[nt]);
    }
  }
  // D layout: VGPR r, lanes0-15 M=r, lanes16-31 M=r+8, N=lane%16
  #pragma unroll
  for (int r = 0; r < 8; ++r) {
    int m = r + 8 * half;
    f16* dst = Xh + (size_t)(Mb + m) * KPAD;
    #pragma unroll
    for (int nt = 0; nt < 3; ++nt) {
      int col = nt * 16 + j16;
      dst[col] = (col < VOC) ? (f16)acc[nt][r] : (f16)0;
    }
    dst[48 + j16] = (f16)0;   // pad 48..63
  }
}

// ---------------------------------------------------------------------------
// K4: Xh[t][b][:] = f16(captions[b][t-1][:]) padded, for t=1..T-1
// ---------------------------------------------------------------------------
__global__ void lstm_packx_kernel(const float* __restrict__ captions,
                                  f16* __restrict__ Xh) {
  size_t idx = (size_t)blockIdx.x * blockDim.x + threadIdx.x; // (T-1)*B*KPAD
  int col = idx & (KPAD - 1);
  size_t row = idx >> 6;                // 0..(255*512-1)
  int t = (int)(row / BATCH) + 1;
  int b = (int)(row % BATCH);
  float v = (col < VOC) ? captions[((size_t)b * TSEQ + (t - 1)) * VOC + col] : 0.f;
  Xh[((size_t)t * BATCH + b) * KPAD + col] = (f16)v;
}

// ---------------------------------------------------------------------------
// K5: pre-gates  G = Xh @ W_ih^T + (b_ih + b_hh), f32.
// M = T*B (8192 waves), N=32 (2 tiles), K=64 (2 k-steps).
// Output stored in WMMA D lane-layout: G[tile][lane][16 floats] so the
// recurrence reloads each tile as one contiguous 64B chunk per lane.
// ---------------------------------------------------------------------------
__global__ void lstm_pregate_kernel(const f16* __restrict__ Xh,
                                    const f16* __restrict__ Wihp,
                                    const float* __restrict__ gbias,
                                    float* __restrict__ G) {
  const int L = threadIdx.x;
  const int j16 = L & 15, half = L >> 4;
  const size_t Mb = (size_t)blockIdx.x * 16;

  v8f acc[2];
  acc[0] = splat8(gbias[j16]);
  acc[1] = splat8(gbias[16 + j16]);

  #pragma unroll
  for (int ks = 0; ks < 2; ++ks) {
    int kb = ks * 32 + half * 8;
    size_t row = Mb + j16;
    v16h A = cat8(*(const v8h*)(Xh + row * KPAD + kb),
                  *(const v8h*)(Xh + row * KPAD + kb + 16));
    #pragma unroll
    for (int nt = 0; nt < 2; ++nt) {
      const f16* tp = Wihp + (size_t)(ks * 2 + nt) * 512 + L * 16;
      v16h Bm = cat8(*(const v8h*)tp, *(const v8h*)(tp + 8));
      acc[nt] = wmma16(A, Bm, acc[nt]);
    }
  }
  float* dst = G + (size_t)blockIdx.x * 512 + L * 16;  // per-lane D layout
  *(v8f*)dst       = acc[0];
  *(v8f*)(dst + 8) = acc[1];
}

// ---------------------------------------------------------------------------
// K6: the recurrence (critical path only). 32 blocks x 1 wave; each wave owns
// 16 batch rows over all 256 steps. Per step: one 64B gate load per lane
// (C operand, L2 resident + prefetch), 2 WMMAs for h·W_hh^T, pointwise cell,
// f16 LDS transpose of h into WMMA-A layout, one b128 store of h_t to the
// history buffer. Softmax head is deferred to the parallel K7.
// ---------------------------------------------------------------------------
__global__ void __launch_bounds__(32)
lstm_recur_kernel(const float* __restrict__ G,
                  const f16* __restrict__ Whhp,
                  f16* __restrict__ Hb) {   // [T*B][8] f16
  const int L = threadIdx.x;
  const int j16 = L & 15, half = L >> 4;
  const int bblk = blockIdx.x;            // batch tile 0..31
  const int bbase = bblk * 16;

  v16h WhhB[2];
  #pragma unroll
  for (int nt = 0; nt < 2; ++nt) {
    const f16* tp = Whhp + (size_t)nt * 512 + L * 16;
    WhhB[nt] = cat8(*(const v8h*)tp, *(const v8h*)(tp + 8));
  }

  float cst[8];
  #pragma unroll
  for (int r = 0; r < 8; ++r) cst[r] = 0.f;

  v16h Ah;                       // packed h_{t-1} in A layout (zero at t=0)
  #pragma unroll
  for (int i = 0; i < 16; ++i) Ah[i] = (f16)0;

  __shared__ f16 hb[16][HID];    // 256B transpose staging

  const bool lowj = (j16 < 8);

  for (int t = 0; t < TSEQ; ++t) {
    // -- contiguous per-lane gate tile (D layout), 64B per lane
    const float* Gt = G + ((size_t)(t * (BATCH / 16) + bblk) * 512) + L * 16;
    v8f g0 = *(const v8f*)Gt;
    v8f g1 = *(const v8f*)(Gt + 8);
    if (t + 1 < TSEQ)
      __builtin_prefetch(Gt + (size_t)(BATCH / 16) * 512, 0, 1);

    // -- gates += h_{t-1} · W_hh^T
    g0 = wmma16(Ah, WhhB[0], g0);
    g1 = wmma16(Ah, WhhB[1], g1);

    // -- pointwise cell; pair (i,f)/(g,o) columns via lane-xor-8 shuffles
    float hnew[8];
    #pragma unroll
    for (int r = 0; r < 8; ++r) {
      float a0 = g0[r], a1 = g1[r];
      float b0 = __shfl_xor(a0, 8, 32);
      float b1 = __shfl_xor(a1, 8, 32);
      float iv = lowj ? a0 : b0;
      float fv = lowj ? b0 : a0;
      float gv = lowj ? a1 : b1;
      float ov = lowj ? b1 : a1;
      iv = sigmoidf_(iv); fv = sigmoidf_(fv); ov = sigmoidf_(ov);
      gv = tanhf_(gv);
      cst[r] = fv * cst[r] + iv * gv;
      hnew[r] = ov * tanhf_(cst[r]);
    }

    // -- f16 transpose through LDS into WMMA A layout
    if (lowj) {
      #pragma unroll
      for (int r = 0; r < 8; ++r) hb[r + 8 * half][j16] = (f16)hnew[r];
    }
    __syncthreads();
    v16h An;
    #pragma unroll
    for (int i = 0; i < 16; ++i) An[i] = (f16)0;
    v8h hv;
    #pragma unroll
    for (int i = 0; i < 8; ++i) hv[i] = (f16)0;
    if (L < 16) {
      hv = *(const v8h*)&hb[L][0];            // one ds_load_b128
      #pragma unroll
      for (int k = 0; k < HID; ++k) An[k] = hv[k];
    }
    __syncthreads();
    Ah = An;

    // -- publish h_t for the parallel output head (off critical path)
    if (L < 16)
      *(v8h*)(Hb + ((size_t)t * BATCH + bbase + L) * HID) = hv;
  }
}

// ---------------------------------------------------------------------------
// K7: output head, fully parallel over all (b,t):
//   probs = softmax(h · W_out^T + b_out).  M = T*B rows (8192 waves),
//   N = 48 (3 tiles, cols >= 37 masked), K = 8 padded to 32.
// ---------------------------------------------------------------------------
__global__ void lstm_outhead_kernel(const f16* __restrict__ Hb,
                                    const f16* __restrict__ Woutp,
                                    const float* __restrict__ obias,
                                    float* __restrict__ out) {
  const int L = threadIdx.x;
  const int j16 = L & 15, half = L >> 4;
  const size_t Mb = (size_t)blockIdx.x * 16;   // row = t*B + b

  v16h WoutB[3];
  #pragma unroll
  for (int nt = 0; nt < 3; ++nt) {
    const f16* tp = Woutp + (size_t)nt * 512 + L * 16;
    WoutB[nt] = cat8(*(const v8h*)tp, *(const v8h*)(tp + 8));
  }

  // A operand: lanes 0-15 carry rows' K=0..7; everything else zero-padded
  v16h A;
  #pragma unroll
  for (int i = 0; i < 16; ++i) A[i] = (f16)0;
  if (L < 16) {
    v8h hv = *(const v8h*)(Hb + (Mb + L) * HID);
    #pragma unroll
    for (int k = 0; k < HID; ++k) A[k] = hv[k];
  }

  v8f l0 = splat8(obias[j16]);
  v8f l1 = splat8(obias[16 + j16]);
  v8f l2 = splat8(obias[32 + j16]);
  l0 = wmma16(A, WoutB[0], l0);
  l1 = wmma16(A, WoutB[1], l1);
  l2 = wmma16(A, WoutB[2], l2);

  const bool v2ok = (32 + j16) < VOC;
  #pragma unroll
  for (int r = 0; r < 8; ++r) {
    float x2 = v2ok ? l2[r] : -1e30f;
    float mx = fmaxf(fmaxf(l0[r], l1[r]), x2);
    #pragma unroll
    for (int s = 1; s < 16; s <<= 1)
      mx = fmaxf(mx, __shfl_xor(mx, s, 16));
    float e0 = __expf(l0[r] - mx);
    float e1 = __expf(l1[r] - mx);
    float e2 = v2ok ? __expf(x2 - mx) : 0.f;
    float sm = e0 + e1 + e2;
    #pragma unroll
    for (int s = 1; s < 16; s <<= 1)
      sm += __shfl_xor(sm, s, 16);
    float inv = 1.0f / sm;

    size_t row = Mb + r + 8 * half;           // t*B + b
    int t = (int)(row / BATCH);
    int b = (int)(row % BATCH);
    float* dst = out + ((size_t)b * TSEQ + t) * VOC;
    dst[j16]      = e0 * inv;
    dst[16 + j16] = e1 * inv;
    if (v2ok) dst[32 + j16] = e2 * inv;
  }
}

// ---------------------------------------------------------------------------
extern "C" void kernel_launch(void* const* d_in, const int* in_sizes, int n_in,
                              void* d_out, int out_size, void* d_ws, size_t ws_size,
                              hipStream_t stream) {
  const float* features = (const float*)d_in[0];
  const float* captions = (const float*)d_in[1];
  const float* W_in  = (const float*)d_in[2];
  const float* b_in  = (const float*)d_in[3];
  const float* W_ih  = (const float*)d_in[4];
  const float* W_hh  = (const float*)d_in[5];
  const float* b_ih  = (const float*)d_in[6];
  const float* b_hh  = (const float*)d_in[7];
  const float* W_out = (const float*)d_in[8];
  const float* b_out = (const float*)d_in[9];
  float* out = (float*)d_out;

  char* ws = (char*)d_ws;
  auto carve = [&](size_t bytes) -> char* {
    char* p = ws;
    ws += (bytes + 255) & ~(size_t)255;
    return p;
  };
  f16*   pooled = (f16*)  carve((size_t)BATCH * CNNC * sizeof(f16));
  f16*   Winp   = (f16*)  carve((size_t)48 * CNNC * sizeof(f16));
  f16*   Wihp   = (f16*)  carve((size_t)4 * 512 * sizeof(f16));
  f16*   Whhp   = (f16*)  carve((size_t)2 * 512 * sizeof(f16));
  f16*   Woutp  = (f16*)  carve((size_t)3 * 512 * sizeof(f16));
  float* gbias  = (float*)carve(GN * sizeof(float));
  float* obias  = (float*)carve(48 * sizeof(float));
  f16*   Xh     = (f16*)  carve((size_t)NROW * KPAD * sizeof(f16));
  float* G      = (float*)carve((size_t)NROW * GN * sizeof(float));
  f16*   Hb     = (f16*)  carve((size_t)NROW * HID * sizeof(f16));

  lstm_pool_kernel<<<(BATCH * CNNC) / 256, 256, 0, stream>>>(features, pooled);
  lstm_pack_kernel<<<1, 256, 0, stream>>>(W_in, W_ih, W_hh, W_out,
                                          b_ih, b_hh, b_out,
                                          Winp, Wihp, Whhp, Woutp, gbias, obias);
  lstm_featgemm_kernel<<<BATCH / 16, 32, 0, stream>>>(pooled, Winp, b_in, Xh);
  lstm_packx_kernel<<<((TSEQ - 1) * BATCH * KPAD) / 256, 256, 0, stream>>>(captions, Xh);
  lstm_pregate_kernel<<<NTILE, 32, 0, stream>>>(Xh, Wihp, gbias, G);
  lstm_recur_kernel<<<BATCH / 16, 32, 0, stream>>>(G, Whhp, Hb);
  lstm_outhead_kernel<<<NTILE, 32, 0, stream>>>(Hb, Woutp, obias, out);
}
